// RWACell_65455301591398
// MI455X (gfx1250) — matrix operand order repaired and checked by reading, not compiled, not run on hardware
//
#include <hip/hip_runtime.h>
#include <hip/hip_bf16.h>

typedef __attribute__((ext_vector_type(16))) __bf16 v16bf;
typedef __attribute__((ext_vector_type(4)))  __bf16 v4bf;
typedef __attribute__((ext_vector_type(8)))  float  v8f;

#define B_ROWS 16384
#define F_DIM  256
#define C_DIM  256
#define K_DIM  512      // F + C (Wu zero-padded to this)
#define BM     32       // rows per workgroup
#define LDSK   520      // padded LDS row stride (bf16 elems): 1040B -> 4-bank shift/row

union Frag { v16bf v; uint4 q[2]; };

__device__ __forceinline__ float fast_tanh(float x) {
  float e = __expf(2.0f * fabsf(x));
  float t = 1.0f - 2.0f / (e + 1.0f);
  return copysignf(t, x);
}

__global__ __launch_bounds__(256) void cvt_bf16(const float* __restrict__ src,
                                                __bf16* __restrict__ dst, int n) {
  int i = blockIdx.x * blockDim.x + threadIdx.x;
  if (i < n) dst[i] = (__bf16)src[i];
}

// Wu [256,256] -> bf16 [256,512] with zeros in k = 256..511 (uniform K for the pipeline)
__global__ __launch_bounds__(256) void cvt_pad_u(const float* __restrict__ src,
                                                 __bf16* __restrict__ dst) {
  int i = blockIdx.x * blockDim.x + threadIdx.x;   // 0 .. 256*512-1
  int row = i >> 9;
  int k   = i & 511;
  dst[i] = (k < 256) ? (__bf16)src[row * 256 + k] : (__bf16)0.0f;
}

__global__ __launch_bounds__(256) void rwa_fused(
    const float* __restrict__ x_t,  const float* __restrict__ n_t,
    const float* __restrict__ d_t,  const float* __restrict__ h_t,
    const float* __restrict__ amax_t,
    const __bf16* __restrict__ WuB, const __bf16* __restrict__ WgB,
    const __bf16* __restrict__ WaB, const __bf16* __restrict__ WdB,
    const float* __restrict__ bu,   const float* __restrict__ bg,
    float* __restrict__ out)
{
  __shared__ __bf16 sA[BM * LDSK];

  const int tid      = threadIdx.x;
  const int blockRow = blockIdx.x * BM;

  // ---- Stage [x | h] block as bf16 into LDS (32 rows x 512 cols) ----
  #pragma unroll
  for (int it = 0; it < 16; ++it) {
    int vec = it * 256 + tid;
    int row = vec >> 7;
    int col = (vec & 127) << 2;
    const float* src = (col < F_DIM)
        ? (x_t + (size_t)(blockRow + row) * F_DIM + col)
        : (h_t + (size_t)(blockRow + row) * C_DIM + (col - F_DIM));
    float4 f = *(const float4*)src;
    v4bf b;
    b.x = (__bf16)f.x; b.y = (__bf16)f.y; b.z = (__bf16)f.z; b.w = (__bf16)f.w;
    *(v4bf*)&sA[row * LDSK + col] = b;
  }
  __syncthreads();

  // ---- Wave tiling: 8 waves = 2 M-tiles x 4 N-groups (64 cols each) ----
  const int lane = tid & 31;
  const int wv   = tid >> 5;
  const int mt   = wv & 1;
  const int n0   = (wv >> 1) * 64;
  const int ll   = lane & 15;
  const int hi   = lane >> 4;

  v8f accU[4], accG[4], accA[4], accD[4];
  {
    v8f z = {};
    #pragma unroll
    for (int jt = 0; jt < 4; ++jt) { accU[jt] = z; accG[jt] = z; accA[jt] = z; accD[jt] = z; }
  }

  const __bf16* srow = &sA[(mt * 16 + ll) * LDSK];

  // per-lane element offsets into any weight matrix (all are [256 x 512] bf16 now)
  int voff[4];
  #pragma unroll
  for (int jt = 0; jt < 4; ++jt) voff[jt] = (n0 + jt * 16 + ll) * K_DIM + hi * 16;

#define LOADA(DST, KS)                                                   \
  { const __bf16* ap = srow + (KS) * 32 + hi * 16;                       \
    (DST).q[0] = *(const uint4*)(ap);                                    \
    (DST).q[1] = *(const uint4*)(ap + 8); }

#define LOADB(BUF, W, KS)                                                \
  { _Pragma("unroll")                                                    \
    for (int jt = 0; jt < 4; ++jt) {                                     \
      const __bf16* wp = (W) + voff[jt] + (KS) * 32;                     \
      BUF[jt].q[0] = *(const uint4*)(wp);                                \
      BUF[jt].q[1] = *(const uint4*)(wp + 8);                            \
    } }

#define WMMA4(ACC, BUF)                                                  \
  { _Pragma("unroll")                                                    \
    for (int jt = 0; jt < 4; ++jt) {                                     \
      ACC[jt] = __builtin_amdgcn_wmma_f32_16x16x32_bf16(                 \
          false, a_cur.v, false, BUF[jt].v, (short)0, ACC[jt], false, false); \
    } }

  Frag a_cur, a_nxt;
  Frag bA[4], bB[4];

  LOADA(a_cur, 0);
  LOADB(bA, WuB, 0);                      // prime the pipeline with U(ks=0)

  #pragma unroll
  for (int ks = 0; ks < 16; ++ks) {
    int ksn = (ks + 1) & 15;              // wrap: last-iter prefetch is dead code after unroll

    LOADB(bB, WgB, ks);                   // prefetch G while computing U
    __builtin_amdgcn_sched_barrier(0);
    WMMA4(accU, bA);
    __builtin_amdgcn_sched_barrier(0);

    LOADB(bA, WaB, ks);                   // prefetch A while computing G
    __builtin_amdgcn_sched_barrier(0);
    WMMA4(accG, bB);
    __builtin_amdgcn_sched_barrier(0);

    LOADB(bB, WdB, ks);                   // prefetch D while computing A
    __builtin_amdgcn_sched_barrier(0);
    WMMA4(accA, bA);
    __builtin_amdgcn_sched_barrier(0);

    LOADB(bA, WuB, ksn);                  // prefetch next U while computing D
    LOADA(a_nxt, ksn);
    __builtin_amdgcn_sched_barrier(0);
    WMMA4(accD, bB);
    __builtin_amdgcn_sched_barrier(0);

    a_cur = a_nxt;
  }

#undef LOADA
#undef LOADB
#undef WMMA4

  // ---- Fused epilogue on accumulator layout: VGPR r -> (M = r + 8*hi, N = ll) ----
  const size_t BC = (size_t)B_ROWS * C_DIM;
  #pragma unroll
  for (int jt = 0; jt < 4; ++jt) {
    int n = n0 + jt * 16 + ll;
    float bun = bu[n];
    float bgn = bg[n];
    #pragma unroll
    for (int r = 0; r < 8; ++r) {
      int row    = blockRow + mt * 16 + hi * 8 + r;
      size_t idx = (size_t)row * C_DIM + n;

      float u   = accU[jt][r] + bun;
      float g   = accG[jt][r] + bgn;
      float at  = accA[jt][r];
      float dec = 1.0f / (1.0f + __expf(-accD[jt][r]));   // sigmoid

      float e_neg      = __expf(-dec);
      float am         = amax_t[idx];
      float a_decay    = am * e_neg;
      float a_new      = fmaxf(a_decay, at);
      float exp_diff   = __expf(am - a_new);
      float exp_scaled = __expf(at - a_new);
      float z          = u * fast_tanh(g);
      float ed         = e_neg * exp_diff;

      float n_new = n_t[idx] * ed + z * exp_scaled;
      float d_new = d_t[idx] * ed + exp_scaled;
      float h_new = fast_tanh(n_new / d_new);

      out[idx]          = n_new;
      out[BC + idx]     = d_new;
      out[2 * BC + idx] = h_new;
      out[3 * BC + idx] = a_new;
    }
  }
}

extern "C" void kernel_launch(void* const* d_in, const int* in_sizes, int n_in,
                              void* d_out, int out_size, void* d_ws, size_t ws_size,
                              hipStream_t stream) {
  const float* x_t  = (const float*)d_in[0];
  const float* n_t  = (const float*)d_in[1];
  const float* d_t  = (const float*)d_in[2];
  const float* h_t  = (const float*)d_in[3];
  const float* amax = (const float*)d_in[4];
  const float* Wu   = (const float*)d_in[5];
  const float* bu   = (const float*)d_in[6];
  const float* Wg   = (const float*)d_in[7];
  const float* bg   = (const float*)d_in[8];
  const float* Wa   = (const float*)d_in[9];
  const float* Wd   = (const float*)d_in[10];

  // bf16 weight staging in workspace (~1 MB total, L2-resident).
  // All four matrices are stored as [256 x 512] bf16 (Wu zero-padded).
  __bf16* ws  = (__bf16*)d_ws;
  __bf16* WuB = ws;
  __bf16* WgB = WuB + 256 * 512;
  __bf16* WaB = WgB + 256 * 512;
  __bf16* WdB = WaB + 256 * 512;

  cvt_pad_u<<<(256 * 512 + 255) / 256, 256, 0, stream>>>(Wu, WuB);
  cvt_bf16<<<(256 * 512 + 255) / 256, 256, 0, stream>>>(Wg, WgB, 256 * 512);
  cvt_bf16<<<(256 * 512 + 255) / 256, 256, 0, stream>>>(Wa, WaB, 256 * 512);
  cvt_bf16<<<(256 * 512 + 255) / 256, 256, 0, stream>>>(Wd, WdB, 256 * 512);

  rwa_fused<<<B_ROWS / BM, 256, 0, stream>>>(
      x_t, n_t, d_t, h_t, amax, WuB, WgB, WaB, WdB, bu, bg, (float*)d_out);
}
